// LSTM_25829933318238
// MI455X (gfx1250) — compile-verified, matrix operand order checked by reading, not run on hardware
//
#include <hip/hip_runtime.h>
#include <stdint.h>
#include <stddef.h>

#define SEQ    512
#define BATCH  128
#define D_IN   1024
#define D_OUT  1024
#define D_XH   2048
#define NTILE  64      // 1024 out cols / 16
#define KSTEPS 64      // 2048 / 32
#define KO     16      // outer K iterations
#define KI     4       // ksteps staged per outer iteration

// halfs per gate in the packed weight buffer
#define GSTR ((size_t)NTILE * KSTEPS * 512)   // 2,097,152 halfs

typedef __attribute__((ext_vector_type(16))) __bf16       v16bf;
typedef __attribute__((ext_vector_type(8)))  __bf16       v8bf;
typedef __attribute__((ext_vector_type(8)))  float        v8f;
typedef __attribute__((ext_vector_type(4)))  unsigned int v4u;
typedef __attribute__((ext_vector_type(4)))  int          v4i;
typedef __attribute__((ext_vector_type(8)))  int          v8i;

#if defined(__has_builtin)
#if __has_builtin(__builtin_amdgcn_tensor_load_to_lds) && \
    __has_builtin(__builtin_amdgcn_s_wait_tensorcnt)
#define USE_TDM 1
#endif
#endif
#ifndef USE_TDM
#define USE_TDM 0
#endif

union V16U { v16bf v; v8bf h[2]; };

static __device__ __forceinline__ v8bf cvt8(v8f x) {
    v8bf r;
#pragma unroll
    for (int i = 0; i < 8; ++i) r[i] = (__bf16)x[i];   // native v_cvt_pk_bf16_f32
    return r;
}

static __device__ __forceinline__ float sigmoidf_(float x) {
    return 1.0f / (1.0f + __expf(-x));
}
static __device__ __forceinline__ float tanhf_(float x) {
    return 1.0f - 2.0f / (__expf(2.0f * x) + 1.0f);
}

// ---------------------------------------------------------------------------
// Prep 1: pack weights fp32 -> bf16 in WMMA B-fragment order.
// Per (gate, ntile j, kstep kk): 512-half block laid as [half(2)][lane(32)][8],
// where lane l covers column n = 16*j + (l&15), koff = (l>=16)?8:0 and
// half selects K subrange {k0+koff..} vs {k0+16+koff..}.
// Block order: g-major, then j, then kk  ==> per (g,j) the kk blocks are
// contiguous, so a KI-block chunk is one contiguous 4 KB run per gate.
// ---------------------------------------------------------------------------
__global__ void __launch_bounds__(256)
pack_weights_kernel(const float* __restrict__ Wf, const float* __restrict__ Wi,
                    const float* __restrict__ Wc, const float* __restrict__ Wo,
                    unsigned short* __restrict__ Wpk) {
    int tid = blockIdx.x * 256 + threadIdx.x;        // 4*64*64*64 = 1,048,576
    int c   = tid & 63;                              // 8-half chunk within block
    int kk  = (tid >> 6)  & 63;
    int j   = (tid >> 12) & 63;
    int g   = (tid >> 18) & 3;
    const float* W = (g == 0) ? Wf : (g == 1) ? Wi : (g == 2) ? Wc : Wo;
    int lane = c & 31;
    int half = c >> 5;
    int n    = j * 16 + (lane & 15);
    int koff = (lane >> 4) << 3;
    const float* src = W + (size_t)n * D_XH + kk * 32 + half * 16 + koff;
    v8bf d;
#pragma unroll
    for (int e = 0; e < 8; ++e) d[e] = (__bf16)src[e];
    *(v8bf*)(void*)(Wpk + (size_t)tid * 8) = d;
}

// Prep 2: zero cell state (fp32) and h0 buffer (bf16) -- ws is poisoned.
__global__ void __launch_bounds__(256)
init_state_kernel(float* __restrict__ c_state, unsigned short* __restrict__ h0) {
    int tid = blockIdx.x * 256 + threadIdx.x;
    if (tid < BATCH * D_OUT) {
        c_state[tid] = 0.0f;
        h0[tid] = 0;
    }
}

#if USE_TDM
// Issue one TDM op: 2D tile = [4 gates] x [KI*512 contiguous halfs] from the
// packed weight buffer into LDS (contiguous). Tracked by TENSORcnt.
static __device__ __forceinline__ void tdm_stage(const unsigned short* gsrc,
                                                 unsigned short* ldst) {
    unsigned lds_off = (unsigned)(uintptr_t)ldst;          // LDS byte offset
    unsigned long long ga = (unsigned long long)(uintptr_t)gsrc;
    v4u g0;
    g0[0] = 1u;                                            // count=1, user D#
    g0[1] = lds_off;                                       // lds_addr
    g0[2] = (unsigned)ga;                                  // global_addr lo
    g0[3] = ((unsigned)(ga >> 32) & 0x01FFFFFFu) | (2u << 30);  // addr hi | type=2
    v8i g1;
    g1[0] = 0x10000;                     // workgroup_mask=0, data_size=1 (2B)
    g1[1] = (int)((KI * 512) << 16);     // tensor_dim0[15:0] << 16   (=2048)
    g1[2] = (int)(4u << 16);             // tensor_dim1 = 4 gates
    g1[3] = (int)((KI * 512) << 16);     // tile_dim0 = 2048 halfs
    g1[4] = 4;                           // tile_dim1 = 4, tile_dim2 = 0
    g1[5] = (int)GSTR;                   // tensor_dim0_stride = inter-gate stride
    g1[6] = 0;
    g1[7] = 0;
    v4i z4; z4[0] = 0; z4[1] = 0; z4[2] = 0; z4[3] = 0;
    v8i z8;
#pragma unroll
    for (int i = 0; i < 8; ++i) z8[i] = 0;
    __builtin_amdgcn_tensor_load_to_lds(g0, g1, z4, z4, z8, 0);
}
#else
// Portable fallback: cooperative global->VGPR->LDS staging, same LDS layout.
static __device__ __forceinline__ void coop_stage(const unsigned short* gsrc,
                                                  unsigned short* ldst, int tid) {
#pragma unroll
    for (int q = 0; q < 4; ++q) {
        int idx = tid + 256 * q;             // 1024 chunks of 16B = 16 KB
        int g = idx >> 8, c = idx & 255;
        *(v8bf*)(void*)(ldst + g * (KI * 512) + c * 8) =
            *(const v8bf*)(const void*)(gsrc + (size_t)g * GSTR + c * 8);
    }
}
#endif

// One ko iteration of the K loop: 4 ksteps against the staged LDS buffer.
// XPART selects the A-matrix source at compile time (tokens fp32 vs h bf16).
template <bool XPART>
static __device__ __forceinline__ void
ko_body(int ko, const unsigned short* __restrict__ ldsbuf,
        const float* __restrict__ xrow, const unsigned short* __restrict__ hrow,
        int koff, int lane, v8f& acc0, v8f& acc1, v8f& acc2, v8f& acc3) {
#pragma unroll
    for (int ki = 0; ki < KI; ++ki) {
        const int kk = ko * KI + ki;
        const int k0 = kk * 32;

        V16U a;
        if (XPART) {
            v8f x0 = *(const v8f*)(xrow + k0 + koff);
            v8f x1 = *(const v8f*)(xrow + k0 + 16 + koff);
            a.h[0] = cvt8(x0);
            a.h[1] = cvt8(x1);
        } else {
            const int hk = k0 - D_IN;
            a.h[0] = *(const v8bf*)(const void*)(hrow + hk + koff);
            a.h[1] = *(const v8bf*)(const void*)(hrow + hk + 16 + koff);
        }

        const unsigned short* base = ldsbuf + ki * 512 + lane * 8;
        V16U b0, b1, b2, b3;
        b0.h[0] = *(const v8bf*)(const void*)(base);
        b0.h[1] = *(const v8bf*)(const void*)(base + 256);
        b1.h[0] = *(const v8bf*)(const void*)(base + 1 * KI * 512);
        b1.h[1] = *(const v8bf*)(const void*)(base + 1 * KI * 512 + 256);
        b2.h[0] = *(const v8bf*)(const void*)(base + 2 * KI * 512);
        b2.h[1] = *(const v8bf*)(const void*)(base + 2 * KI * 512 + 256);
        b3.h[0] = *(const v8bf*)(const void*)(base + 3 * KI * 512);
        b3.h[1] = *(const v8bf*)(const void*)(base + 3 * KI * 512 + 256);

        acc0 = __builtin_amdgcn_wmma_f32_16x16x32_bf16(false, a.v, false, b0.v, (short)0, acc0, false, false);
        acc1 = __builtin_amdgcn_wmma_f32_16x16x32_bf16(false, a.v, false, b1.v, (short)0, acc1, false, false);
        acc2 = __builtin_amdgcn_wmma_f32_16x16x32_bf16(false, a.v, false, b2.v, (short)0, acc2, false, false);
        acc3 = __builtin_amdgcn_wmma_f32_16x16x32_bf16(false, a.v, false, b3.v, (short)0, acc3, false, false);
    }
}

// ---------------------------------------------------------------------------
// One LSTM timestep. Grid: 64 blocks (16 output cols each) x 8 waves (16
// batch rows each). Weights staged per-block into LDS (TDM double buffer),
// each wave holds all 4 gate accumulators -> cell update is per-lane math.
// ---------------------------------------------------------------------------
__global__ void __launch_bounds__(256)
lstm_step_kernel(const float* __restrict__ tokens,          // [SEQ][BATCH][D_IN]
                 const unsigned short* __restrict__ Wpk,    // packed bf16 weights
                 const float* __restrict__ bfp, const float* __restrict__ bip,
                 const float* __restrict__ bcp, const float* __restrict__ bop,
                 const unsigned short* __restrict__ h_in,   // [BATCH][D_OUT] bf16
                 unsigned short* __restrict__ h_out,        // [BATCH][D_OUT] bf16
                 float* __restrict__ c_state,               // [BATCH][D_OUT]
                 float* __restrict__ out,                   // [SEQ][BATCH][D_OUT]
                 int t) {
    __shared__ unsigned short ldsW[2][4 * KI * 512];        // 2 x 16 KB

    const int j    = blockIdx.x;
    const int tid  = threadIdx.x;
    const int lane = tid & 31;
    const int w    = tid >> 5;
    const int m    = lane & 15;
    const int koff = (lane >> 4) << 3;
    const int n    = j * 16 + m;

    float vbf = bfp[n], vbi = bip[n], vbc = bcp[n], vbo = bop[n];
    v8f acc0, acc1, acc2, acc3;
#pragma unroll
    for (int r = 0; r < 8; ++r) { acc0[r] = vbf; acc1[r] = vbi; acc2[r] = vbc; acc3[r] = vbo; }

    const float*          xrow = tokens + ((size_t)t * BATCH + (16 * w + m)) * D_IN;
    const unsigned short* hrow = h_in + (size_t)(16 * w + m) * D_OUT;
    const unsigned short* wsrc = Wpk + (size_t)j * KSTEPS * 512;   // gate 0, kk 0

    // stage chunk 0
#if USE_TDM
    if (w == 0) { tdm_stage(wsrc, &ldsW[0][0]); __builtin_amdgcn_s_wait_tensorcnt(0); }
#else
    coop_stage(wsrc, &ldsW[0][0], tid);
#endif
    __syncthreads();

    // x phase: ko 0..7 (A from tokens, fp32 -> bf16)
    for (int ko = 0; ko < KO / 2; ++ko) {
#if USE_TDM
        if (w == 0) tdm_stage(wsrc + (size_t)(ko + 1) * KI * 512, &ldsW[(ko & 1) ^ 1][0]);
#else
        coop_stage(wsrc + (size_t)(ko + 1) * KI * 512, &ldsW[(ko & 1) ^ 1][0], tid);
#endif
        ko_body<true>(ko, &ldsW[ko & 1][0], xrow, hrow, koff, lane, acc0, acc1, acc2, acc3);
#if USE_TDM
        if (w == 0) __builtin_amdgcn_s_wait_tensorcnt(0);
#endif
        __syncthreads();
    }

    // h phase: ko 8..14 (A from h_in, bf16 direct), last chunk still staged
    for (int ko = KO / 2; ko < KO - 1; ++ko) {
#if USE_TDM
        if (w == 0) tdm_stage(wsrc + (size_t)(ko + 1) * KI * 512, &ldsW[(ko & 1) ^ 1][0]);
#else
        coop_stage(wsrc + (size_t)(ko + 1) * KI * 512, &ldsW[(ko & 1) ^ 1][0], tid);
#endif
        ko_body<false>(ko, &ldsW[ko & 1][0], xrow, hrow, koff, lane, acc0, acc1, acc2, acc3);
#if USE_TDM
        if (w == 0) __builtin_amdgcn_s_wait_tensorcnt(0);
#endif
        __syncthreads();
    }

    // final ko = 15: nothing left to stage
    ko_body<false>(KO - 1, &ldsW[(KO - 1) & 1][0], xrow, hrow, koff, lane,
                   acc0, acc1, acc2, acc3);

    // ---- LSTM cell update: per-lane, all 4 gates resident in registers ----
    // C/D layout: VGPR r holds (M = r + (lane>=16 ? 8 : 0), N = lane&15)
    const int rbase = 16 * w + ((lane >> 4) << 3);
#pragma unroll
    for (int r = 0; r < 8; ++r) {
        const int b = rbase + r;
        float fg = sigmoidf_(acc0[r]);
        float ig = sigmoidf_(acc1[r]);
        float gg = tanhf_(acc2[r]);
        float og = sigmoidf_(acc3[r]);
        const size_t cidx = (size_t)b * D_OUT + n;
        float c_new = c_state[cidx] * fg + ig * gg;
        c_state[cidx] = c_new;
        float h = og * tanhf_(c_new);
        out[((size_t)t * BATCH + b) * D_OUT + n] = h;
        union { __bf16 h; unsigned short s; } u;
        u.h = (__bf16)h;
        h_out[cidx] = u.s;
    }
}

// ---------------------------------------------------------------------------
extern "C" void kernel_launch(void* const* d_in, const int* in_sizes, int n_in,
                              void* d_out, int out_size, void* d_ws, size_t ws_size,
                              hipStream_t stream) {
    const float* tokens = (const float*)d_in[0];
    const float* W_f = (const float*)d_in[1];
    const float* b_f = (const float*)d_in[2];
    const float* W_i = (const float*)d_in[3];
    const float* b_i = (const float*)d_in[4];
    const float* W_c = (const float*)d_in[5];
    const float* b_c = (const float*)d_in[6];
    const float* W_o = (const float*)d_in[7];
    const float* b_o = (const float*)d_in[8];
    float* out = (float*)d_out;

    // ws: [0,16MB) packed bf16 weights; h ping-pong (2 x 256KB bf16); c (512KB)
    char* ws = (char*)d_ws;
    unsigned short* Wpk   = (unsigned short*)ws;
    unsigned short* hbuf0 = (unsigned short*)(ws + ((size_t)16 << 20));
    unsigned short* hbuf1 = hbuf0 + (size_t)BATCH * D_OUT;
    float*          c_st  = (float*)(hbuf1 + (size_t)BATCH * D_OUT);

    pack_weights_kernel<<<4096, 256, 0, stream>>>(W_f, W_i, W_c, W_o, Wpk);
    init_state_kernel<<<(BATCH * D_OUT + 255) / 256, 256, 0, stream>>>(c_st, hbuf0);

    for (int t = 0; t < SEQ; ++t) {
        const unsigned short* hin = (t & 1) ? hbuf1 : hbuf0;
        unsigned short*      hout = (t & 1) ? hbuf0 : hbuf1;
        lstm_step_kernel<<<NTILE, 256, 0, stream>>>(tokens, Wpk,
                                                    b_f, b_i, b_c, b_o,
                                                    hin, hout, c_st, out, t);
    }
}
